// RingConv2dSimple_18476949307606
// MI455X (gfx1250) — matrix operand order repaired and checked by reading, not compiled, not run on hardware
//
#include <hip/hip_runtime.h>

typedef _Float16 half_t;
typedef __attribute__((ext_vector_type(16))) _Float16 v16h;
typedef __attribute__((ext_vector_type(8)))  _Float16 v8h;
typedef __attribute__((ext_vector_type(4)))  _Float16 v4h;
typedef __attribute__((ext_vector_type(8)))  float    v8f;

#define B_     4
#define CIN_   32
#define COUT_  64
#define HW_    32
#define C4_    128      // 4*CIN feature channels
#define N_     128      // 2*COUT (x-part | y-part)
#define KL_    9
#define M_TOT  4096     // B*H*W
#define KSTEPS 36       // (KL_*C4_)/32
#define LDSS   40       // padded LDS row stride in halves (80 B)

// ---------------- feature kernel: x -> [cos p, sin p, cos 3p, sin 3p] -------
__global__ void ring_features(const float* __restrict__ x, half_t* __restrict__ F) {
  int idx = blockIdx.x * blockDim.x + threadIdx.x;     // over B*CIN*H*W
  if (idx >= B_ * CIN_ * HW_ * HW_) return;
  int w   = idx & 31;
  int h   = (idx >> 5) & 31;
  int cin = (idx >> 10) & (CIN_ - 1);
  int b   = idx >> 15;
  float v = x[idx];
  float c = __cosf(v), s = __sinf(v);
  float c3 = c * (4.0f * c * c - 3.0f);
  float s3 = s * (3.0f - 4.0f * s * s);
  v4h o;
  o[0] = (half_t)c; o[1] = (half_t)s; o[2] = (half_t)c3; o[3] = (half_t)s3;
  // F layout: [b][h][w][C4_], c4 = 4*cin + f  (channel-last for contiguous GEMM-A rows)
  *(v4h*)&F[(size_t)((b * 1024 + h * 32 + w) * C4_ + 4 * cin)] = o;
}

// ---------------- weight kernel: probe/out_w -> Wm[kl][n][c4] (f16) --------
__global__ void ring_weights(const float* __restrict__ probe,
                             const float* __restrict__ outw,
                             half_t* __restrict__ Wm) {
  int idx = blockIdx.x * blockDim.x + threadIdx.x;     // over CIN*COUT*9
  if (idx >= CIN_ * COUT_ * KL_) return;
  int kl  = idx % KL_;
  int o   = (idx / KL_) % COUT_;
  int cin = idx / (KL_ * COUT_);
  float th = probe[idx], wv = outw[idx];
  float cth = __cosf(th), sth = __sinf(th);
  float c3  = cth * (4.0f * cth * cth - 3.0f);
  float s3  = sth * (3.0f - 4.0f * sth * sth);
  float cw  = __cosf(wv), sw = __sinf(wv);
  float coef[4] = {0.75f * cth, 0.75f * sth, 0.25f * c3, 0.25f * s3};
  size_t basex = (size_t)(kl * N_ + o) * C4_ + 4 * cin;          // x-part: n = o
  size_t basey = (size_t)(kl * N_ + 64 + o) * C4_ + 4 * cin;     // y-part: n = o+64
#pragma unroll
  for (int f = 0; f < 4; ++f) {
    Wm[basex + f] = (half_t)(cw * coef[f]);
    Wm[basey + f] = (half_t)(sw * coef[f]);
  }
}

static __device__ inline v16h cat8(v8h lo, v8h hi) {
  return __builtin_shufflevector(lo, hi, 0,1,2,3,4,5,6,7,8,9,10,11,12,13,14,15);
}

// ---------------- implicit-GEMM conv with WMMA + fused atan2 ---------------
__global__ __launch_bounds__(256) void ring_gemm(const half_t* __restrict__ F,
                                                 const half_t* __restrict__ Wm,
                                                 float* __restrict__ out) {
  __shared__ __align__(16) half_t As[128 * LDSS];   // A tile: 128 rows x 32 K
  __shared__ __align__(16) half_t Bs[128 * LDSS];   // B tile (K-contig per col): 128 cols x 32 K

  const int tid   = threadIdx.x;
  const int lane  = tid & 31;
  const int wave  = tid >> 5;        // 0..7
  const int waveM = wave & 3;        // 4 wave-rows  -> 4*32 = 128 M
  const int waveN = wave >> 2;       // 2 wave-cols  -> 2*32 = 64 cols, paired with +64
  const int mBase = blockIdx.x * 128;

  const int lrow  = lane & 15;
  const int khalf = lane >> 4;       // selects the per-lane-group K half

  // staging assignment: each thread copies 16 halves of one row
  const int copyRow  = tid >> 1;        // 0..127
  const int copyHalf = (tid & 1) * 16;  // 0 or 16 halves

  const int am = mBase + copyRow;
  const int ab = am >> 10;
  const int ah = (am >> 5) & 31;
  const int aw = am & 31;

  v8f acc[2][4];
#pragma unroll
  for (int i = 0; i < 2; ++i)
#pragma unroll
    for (int j = 0; j < 4; ++j) acc[i][j] = (v8f){};

  for (int ks = 0; ks < KSTEPS; ++ks) {
    const int kl  = ks >> 2;             // kernel tap 0..8
    const int c4b = (ks & 3) * 32;       // 32-channel slice
    const int y   = ah + (kl / 3) - 1;
    const int x   = aw + (kl % 3) - 1;

    // ---- stage A tile (implicit im2col row; zero-pad out-of-bounds) ----
    v8h* dstA = (v8h*)&As[copyRow * LDSS + copyHalf];
    if ((unsigned)y < 32u && (unsigned)x < 32u) {
      const v8h* srcA = (const v8h*)&F[(size_t)((ab * 1024 + y * 32 + x) * C4_ + c4b + copyHalf)];
      dstA[0] = srcA[0];
      dstA[1] = srcA[1];
    } else {
      v8h z = (v8h){};
      dstA[0] = z;
      dstA[1] = z;
    }
    // ---- stage B tile (already n-major, K contiguous) ----
    {
      const v8h* srcB = (const v8h*)&Wm[(size_t)((kl * N_ + copyRow) * C4_ + c4b + copyHalf)];
      v8h* dstB = (v8h*)&Bs[copyRow * LDSS + copyHalf];
      dstB[0] = srcB[0];
      dstB[1] = srcB[1];
    }
    __syncthreads();

    // ---- A fragments: lanes 0-15 hold K{0..7,16..23}; lanes 16-31 K{8..15,24..31} ----
    v16h afrag[2];
#pragma unroll
    for (int mf = 0; mf < 2; ++mf) {
      const int m  = waveM * 32 + mf * 16 + lrow;
      const int k0 = khalf * 8;
      v8h alo = *(const v8h*)&As[m * LDSS + k0];
      v8h ahi = *(const v8h*)&As[m * LDSS + k0 + 16];
      afrag[mf] = cat8(alo, ahi);
    }
    // ---- B fragments: col = lane%16, lanes 0-15 K{0..7,16..23}; lanes 16-31 K{8..15,24..31} ----
    v16h bfrag[4];
#pragma unroll
    for (int nf = 0; nf < 4; ++nf) {
      const int n  = waveN * 32 + (nf & 1) * 16 + (nf >> 1) * 64 + lrow;
      const int k0 = khalf * 8;
      v8h blo = *(const v8h*)&Bs[n * LDSS + k0];
      v8h bhi = *(const v8h*)&Bs[n * LDSS + k0 + 16];
      bfrag[nf] = cat8(blo, bhi);
    }

#pragma unroll
    for (int mf = 0; mf < 2; ++mf)
#pragma unroll
      for (int nf = 0; nf < 4; ++nf)
        acc[mf][nf] = __builtin_amdgcn_wmma_f32_16x16x32_f16(
            false, afrag[mf], false, bfrag[nf], (short)0, acc[mf][nf], false, false);

    __syncthreads();
  }

  // ---- epilogue: atan2(y-part, x-part); nf pairs (p, p+2) are (x, y) ----
#pragma unroll
  for (int mf = 0; mf < 2; ++mf) {
#pragma unroll
    for (int p = 0; p < 2; ++p) {
      v8f vx = acc[mf][p];
      v8f vy = acc[mf][p + 2];
      const int o    = waveN * 32 + p * 16 + lrow;                 // output channel 0..63
      const int mrow = mBase + waveM * 32 + mf * 16 + khalf * 8;   // C/D layout: M = r + 8*(lane/16)
#pragma unroll
      for (int r = 0; r < 8; ++r) {
        const int m  = mrow + r;
        const int b  = m >> 10;
        const int hw = m & 1023;
        out[(size_t)(b * COUT_ + o) * 1024 + hw] = atan2f(vy[r], vx[r]);
      }
    }
  }
}

extern "C" void kernel_launch(void* const* d_in, const int* in_sizes, int n_in,
                              void* d_out, int out_size, void* d_ws, size_t ws_size,
                              hipStream_t stream) {
  const float* x     = (const float*)d_in[0];
  const float* probe = (const float*)d_in[1];
  const float* outw  = (const float*)d_in[2];
  float* out = (float*)d_out;

  half_t* F  = (half_t*)d_ws;                               // 4*32*32*128 halves = 1 MB
  half_t* Wm = F + (size_t)B_ * HW_ * HW_ * C4_;            // 9*128*128 halves = 288 KB

  {
    int n = B_ * CIN_ * HW_ * HW_;
    ring_features<<<(n + 255) / 256, 256, 0, stream>>>(x, F);
  }
  {
    int n = CIN_ * COUT_ * KL_;
    ring_weights<<<(n + 255) / 256, 256, 0, stream>>>(probe, outw, Wm);
  }
  ring_gemm<<<M_TOT / 128, 256, 0, stream>>>(F, Wm, out);
}
